// Graph_DDPM_net_model_67869073211789
// MI455X (gfx1250) — matrix operand order repaired and checked by reading, not compiled, not run on hardware
//
#include <hip/hip_runtime.h>
#include <math.h>

// ---------------------------------------------------------------------------
// Graph U-Net forward for MI455X (gfx1250, wave32).
// GEMMs: v_wmma_f32_16x16x32_bf16, 4 column tiles per wave (A-fragment reuse),
// with global_prefetch on the streaming A operand. Scatter: float4 gather +
// global f32 atomics (L2-resident: node features <= 100MB << 192MB L2).
// ---------------------------------------------------------------------------

typedef __bf16 bf16_t;
typedef __attribute__((ext_vector_type(16))) __bf16 v16bf;
typedef __attribute__((ext_vector_type(8)))  float  v8f;

#define ELT_BLOCK 256

__device__ __forceinline__ float silu_f(float x) { return x / (1.0f + expf(-x)); }

// ------------------------- small utility kernels ---------------------------

__global__ void k_te(const int* __restrict__ t, float* __restrict__ te) {
  int k = threadIdx.x;
  if (k < 50) {
    float tf = (float)(*t);
    float w = powf(10000.0f, -0.04f * (float)k);   // 10000^{-2*(2k)/100}
    te[2 * k]     = sinf(tf * w);
    te[2 * k + 1] = cosf(tf * w);
  }
}

__global__ void k_deg(const int* __restrict__ dst, float* __restrict__ deg, int E) {
  int e = blockIdx.x * blockDim.x + threadIdx.x;
  if (e < E) atomicAdd(&deg[dst[e]], 1.0f);
}

__global__ void k_rdeg(float* __restrict__ deg, int N) {
  int i = blockIdx.x * blockDim.x + threadIdx.x;
  if (i < N) deg[i] = 1.0f / fmaxf(deg[i], 1.0f);
}

// te_mlp: v1 = silu(W1 @ te + b1); v = W2 @ v1 + b2.  One block, a <= 256.
__global__ void k_temlp(const float* __restrict__ te,
                        const float* __restrict__ w1, const float* __restrict__ b1,
                        const float* __restrict__ w2, const float* __restrict__ b2,
                        float* __restrict__ v, int a) {
  __shared__ float v1[256];
  int j = threadIdx.x;
  if (j < a) {
    float s = 0.0f;
    #pragma unroll 4
    for (int i = 0; i < 100; ++i) s += w1[j * 100 + i] * te[i];
    v1[j] = silu_f(s + b1[j]);
  }
  __syncthreads();
  if (j < a) {
    float s = 0.0f;
    for (int i = 0; i < a; ++i) s += w2[j * a + i] * v1[i];
    v[j] = s + b2[j];
  }
}

// Sum / sum-of-squares of (concat(h1,h2)[n,c] + v[c]) over all N*(C1+C2) elems.
__global__ void k_reduce(const float* __restrict__ h1, int C1,
                         const float* __restrict__ h2, int C2,
                         const float* __restrict__ v, float* __restrict__ stats, int N) {
  int C = C1 + C2;
  size_t total = (size_t)N * C;
  float s = 0.0f, ss = 0.0f;
  for (size_t idx = (size_t)blockIdx.x * blockDim.x + threadIdx.x; idx < total;
       idx += (size_t)gridDim.x * blockDim.x) {
    int c = (int)(idx % C);
    size_t n = idx / C;
    float x = ((c < C1) ? h1[n * C1 + c] : h2[n * C2 + (c - C1)]) + v[c];
    s += x; ss += x * x;
  }
  __shared__ float sh[ELT_BLOCK], sq[ELT_BLOCK];
  sh[threadIdx.x] = s; sq[threadIdx.x] = ss;
  __syncthreads();
  for (int o = ELT_BLOCK / 2; o > 0; o >>= 1) {
    if (threadIdx.x < o) { sh[threadIdx.x] += sh[threadIdx.x + o]; sq[threadIdx.x] += sq[threadIdx.x + o]; }
    __syncthreads();
  }
  if (threadIdx.x == 0) { atomicAdd(&stats[0], sh[0]); atomicAdd(&stats[1], sq[0]); }
}

// Graph-LN: hn = (x+v - mu)/(sd+eps)*g + b ; also emit bf16 copy for WMMA A.
__global__ void k_norm(const float* __restrict__ h1, int C1,
                       const float* __restrict__ h2, int C2,
                       const float* __restrict__ v, const float* __restrict__ g,
                       const float* __restrict__ b, const float* __restrict__ stats,
                       float* __restrict__ hn, bf16_t* __restrict__ hnbf, int N) {
  int C = C1 + C2;
  size_t total = (size_t)N * C;
  float cnt = (float)total;
  float mu = stats[0] / cnt;
  float var = fmaxf(stats[1] / cnt - mu * mu, 0.0f);
  float inv = 1.0f / (sqrtf(var) + 1e-5f);
  for (size_t idx = (size_t)blockIdx.x * blockDim.x + threadIdx.x; idx < total;
       idx += (size_t)gridDim.x * blockDim.x) {
    int c = (int)(idx % C);
    size_t n = idx / C;
    float x = ((c < C1) ? h1[n * C1 + c] : h2[n * C2 + (c - C1)]) + v[c];
    float y = (x - mu) * inv * g[c] + b[c];
    hn[idx] = y;
    hnbf[idx] = (bf16_t)y;
  }
}

// Generic edge scatter-add (used only for C=3): agg[dst[e], c] += feat[src[e], c].
__global__ void k_scatter(const int* __restrict__ src, const int* __restrict__ dst,
                          const float* __restrict__ feat, float* __restrict__ agg,
                          int E, int C) {
  size_t total = (size_t)E * C;
  for (size_t idx = (size_t)blockIdx.x * blockDim.x + threadIdx.x; idx < total;
       idx += (size_t)gridDim.x * blockDim.x) {
    int c = (int)(idx % C);
    size_t e = idx / C;
    atomicAdd(&agg[(size_t)dst[e] * C + c], feat[(size_t)src[e] * C + c]);
  }
}

// Vectorized edge scatter-add for C % 4 == 0: one float4 gather + 4 atomics.
__global__ void k_scatter4(const int* __restrict__ src, const int* __restrict__ dst,
                           const float* __restrict__ feat, float* __restrict__ agg,
                           int E, int C) {
  int C4 = C >> 2;
  size_t total = (size_t)E * C4;
  for (size_t idx = (size_t)blockIdx.x * blockDim.x + threadIdx.x; idx < total;
       idx += (size_t)gridDim.x * blockDim.x) {
    int q = (int)(idx % C4);
    size_t e = idx / C4;
    const float4 val = ((const float4*)(feat + (size_t)src[e] * C))[q];
    float* base = agg + (size_t)dst[e] * C + q * 4;
    atomicAdd(base + 0, val.x);
    atomicAdd(base + 1, val.y);
    atomicAdd(base + 2, val.z);
    atomicAdd(base + 3, val.w);
  }
}

// agg (optionally row-scaled by rdeg for mean aggregation) -> bf16.
__global__ void k_aggbf(const float* __restrict__ agg, const float* __restrict__ rdeg,
                        bf16_t* __restrict__ out, int N, int C) {
  size_t total = (size_t)N * C;
  for (size_t idx = (size_t)blockIdx.x * blockDim.x + threadIdx.x; idx < total;
       idx += (size_t)gridDim.x * blockDim.x) {
    size_t n = idx / C;
    float s = rdeg ? rdeg[n] : 1.0f;
    out[idx] = (bf16_t)(agg[idx] * s);
  }
}

__global__ void k_cvt(const float* __restrict__ src, bf16_t* __restrict__ dst, size_t n) {
  for (size_t idx = (size_t)blockIdx.x * blockDim.x + threadIdx.x; idx < n;
       idx += (size_t)gridDim.x * blockDim.x)
    dst[idx] = (bf16_t)src[idx];
}

// ----------------------------- WMMA GEMM -----------------------------------
// out[n, co] = silu( (A1@W1^T)[n,co] + bias[co] + (A2@W2^T)[n,co] )
// A1/A2: N x Cin bf16 row-major.  W1/W2: Cout x Cin bf16 row-major.
// Each wave owns 16 rows and sweeps 4 column tiles (64 cols), reusing its A
// fragments across tiles: 8 v_wmma per k-step, A traffic amortized 4x.
// Block = 4 waves = 64 rows. Requires Cin%32==0, Cout%64==0.
__global__ __launch_bounds__(128)
void k_gemm_wmma(const bf16_t* __restrict__ A1, const bf16_t* __restrict__ A2,
                 const bf16_t* __restrict__ W1, const bf16_t* __restrict__ W2,
                 const float* __restrict__ bias, float* __restrict__ out,
                 int N, int Cin, int Cout) {
  const int lane = threadIdx.x & 31;
  const int wave = threadIdx.x >> 5;
  const int row0 = blockIdx.x * 64 + wave * 16;   // 16 rows per wave
  const int col0 = blockIdx.y * 64;               // 4 x 16 cols per wave
  if (row0 >= N) return;                          // wave-uniform guard

  const int m    = lane & 15;
  const int hi   = lane >> 4;   // half-wave id
  const int kloA = hi * 8;      // A: lanes<16 -> K {0..7,16..23}; lanes>=16 -> {8..15,24..31}
  const int kloB = hi * 8;      // B: lanes<16 -> K {0..7,16..23}; lanes>=16 -> {8..15,24..31}

  const bf16_t* arow1 = A1 + (size_t)(row0 + m) * Cin;
  const bf16_t* arow2 = A2 + (size_t)(row0 + m) * Cin;
  const bf16_t* wrow[4][2];
  #pragma unroll
  for (int j = 0; j < 4; ++j) {
    wrow[j][0] = W1 + (size_t)(col0 + j * 16 + m) * Cin;
    wrow[j][1] = W2 + (size_t)(col0 + j * 16 + m) * Cin;
  }

  v8f acc[4] = {};
  for (int k0 = 0; k0 < Cin; k0 += 32) {
    if (k0 + 32 < Cin) {                          // prefetch next A strip
      __builtin_prefetch(arow1 + k0 + 32, 0, 3);
      __builtin_prefetch(arow2 + k0 + 32, 0, 3);
    }
    v16bf a1, a2;
    #pragma unroll
    for (int h = 0; h < 8; ++h) {
      a1[h]     = arow1[k0 + kloA + h];
      a1[h + 8] = arow1[k0 + 16 + kloA + h];
      a2[h]     = arow2[k0 + kloA + h];
      a2[h + 8] = arow2[k0 + 16 + kloA + h];
    }
    #pragma unroll
    for (int j = 0; j < 4; ++j) {
      v16bf b1, b2;
      #pragma unroll
      for (int h = 0; h < 16; ++h) {
        b1[h] = wrow[j][0][k0 + kloB + h + (h & 8)];
        b2[h] = wrow[j][1][k0 + kloB + h + (h & 8)];
      }
      acc[j] = __builtin_amdgcn_wmma_f32_16x16x32_bf16(false, a1, false, b1, (short)0, acc[j], false, false);
      acc[j] = __builtin_amdgcn_wmma_f32_16x16x32_bf16(false, a2, false, b2, (short)0, acc[j], false, false);
    }
  }

  const int rbase = row0 + hi * 8;                // D: vgpr r -> M = r + 8*hi, N = lane%16
  #pragma unroll
  for (int j = 0; j < 4; ++j) {
    const int cidx = col0 + j * 16 + m;
    const float bc = bias[cidx];
    #pragma unroll
    for (int r = 0; r < 8; ++r) {
      float y = silu_f(acc[j][r] + bc);
      out[(size_t)(rbase + r) * Cout + cidx] = y;
    }
  }
}

// Scalar GEMM for tiny shapes (down0: Cin=3; last: Cout=3).
__global__ void k_gemm_scalar(const float* __restrict__ A1, const float* __restrict__ A2,
                              const float* __restrict__ rdeg,
                              const float* __restrict__ W1, const float* __restrict__ bias,
                              const float* __restrict__ W2, float* __restrict__ out,
                              int N, int Cin, int Cout, int act) {
  size_t total = (size_t)N * Cout;
  for (size_t idx = (size_t)blockIdx.x * blockDim.x + threadIdx.x; idx < total;
       idx += (size_t)gridDim.x * blockDim.x) {
    size_t n = idx / Cout;
    int co = (int)(idx % Cout);
    float sc = rdeg ? rdeg[n] : 1.0f;
    float s = bias[co];
    for (int c = 0; c < Cin; ++c)
      s += A1[n * Cin + c] * sc * W1[co * Cin + c] + A2[n * Cin + c] * W2[co * Cin + c];
    out[idx] = act ? silu_f(s) : s;
  }
}

// ------------------------------- host side ---------------------------------

static inline int gblk(size_t tot) {
  size_t b = (tot + ELT_BLOCK - 1) / ELT_BLOCK;
  if (b > (size_t)1048576) b = 1048576;   // grid-stride covers remainder
  return (int)b;
}

extern "C" void kernel_launch(void* const* d_in, const int* in_sizes, int n_in,
                              void* d_out, int out_size, void* d_ws, size_t ws_size,
                              hipStream_t stream) {
  (void)out_size; (void)ws_size;
  if (n_in < 60) return;

  const float* x  = (const float*)d_in[0];
  const int*   ei = (const int*)d_in[1];
  const int N = in_sizes[0] / 3;
  const int E = in_sizes[1] / 2;
  const int* srcIdx = ei;
  const int* dstIdx = ei + E;
  const int* tptr = (const int*)d_in[59];

  // Param leaves flattened in setup_inputs() insertion order:
  // down{0,1,2}: te_w1,te_b1,te_w2,te_b2,ln_g,ln_b,wl,bl,wr
  // up{0,1,2}:   te_w1,te_b1,te_w2,te_b2,ln_g,ln_b,wrel,brel,wroot
  // last:        wrel,brel,wroot
  const float* P[57];
  for (int i = 0; i < 57; ++i) P[i] = (const float*)d_in[2 + i];

  // ---- workspace carve (≈ 490 MB) ----
  char* wp = (char*)d_ws;
  auto carve = [&](size_t bytes) { char* p = wp; wp += (bytes + 255) & ~(size_t)255; return p; };
  float*  stats = (float*)carve(64);
  float*  te    = (float*)carve(512);
  float*  tv    = (float*)carve(1024);
  float*  rdeg  = (float*)carve((size_t)N * 4);
  float*  hcur  = (float*)carve((size_t)N * 256 * 4);
  float*  hn    = (float*)carve((size_t)N * 256 * 4);
  float*  agg   = (float*)carve((size_t)N * 256 * 4);
  bf16_t* hnbf  = (bf16_t*)carve((size_t)N * 256 * 2);
  bf16_t* aggbf = (bf16_t*)carve((size_t)N * 256 * 2);
  bf16_t* wbf1  = (bf16_t*)carve((size_t)256 * 256 * 2);
  bf16_t* wbf2  = (bf16_t*)carve((size_t)256 * 256 * 2);
  float*  skip0 = (float*)carve((size_t)N * 64 * 4);
  float*  skip1 = (float*)carve((size_t)N * 128 * 4);

  // ---- degree (mean aggregation) and time embedding ----
  hipMemsetAsync(rdeg, 0, (size_t)N * 4, stream);
  k_deg<<<(E + 255) / 256, 256, 0, stream>>>(dstIdx, rdeg, E);
  k_rdeg<<<(N + 255) / 256, 256, 0, stream>>>(rdeg, N);
  k_te<<<1, 64, 0, stream>>>(tptr, te);

  auto run_pre = [&](const float* h1, int C1, const float* h2, int C2, const float* const* p) {
    int C = C1 + C2;
    k_temlp<<<1, 256, 0, stream>>>(te, p[0], p[1], p[2], p[3], tv, C);
    hipMemsetAsync(stats, 0, 2 * sizeof(float), stream);
    size_t tot = (size_t)N * C;
    int rb = gblk(tot); if (rb > 2048) rb = 2048;   // limit atomic contention
    k_reduce<<<rb, ELT_BLOCK, 0, stream>>>(h1, C1, h2, C2, tv, stats, N);
    k_norm<<<gblk(tot), ELT_BLOCK, 0, stream>>>(h1, C1, h2, C2, tv, p[4], p[5], stats, hn, hnbf, N);
  };

  auto run_scatter = [&](const float* feat, int C) {
    hipMemsetAsync(agg, 0, (size_t)N * C * 4, stream);
    if ((C & 3) == 0)
      k_scatter4<<<gblk((size_t)E * (C >> 2)), ELT_BLOCK, 0, stream>>>(srcIdx, dstIdx, feat, agg, E, C);
    else
      k_scatter<<<gblk((size_t)E * C), ELT_BLOCK, 0, stream>>>(srcIdx, dstIdx, feat, agg, E, C);
  };

  auto run_gemm = [&](const float* w1, const float* w2, const float* bias,
                      int Cin, int Cout, bool mean) {
    k_aggbf<<<gblk((size_t)N * Cin), ELT_BLOCK, 0, stream>>>(agg, mean ? rdeg : (const float*)nullptr,
                                                             aggbf, N, Cin);
    size_t wn = (size_t)Cout * Cin;
    k_cvt<<<gblk(wn), ELT_BLOCK, 0, stream>>>(w1, wbf1, wn);
    k_cvt<<<gblk(wn), ELT_BLOCK, 0, stream>>>(w2, wbf2, wn);
    dim3 grid((N + 63) / 64, Cout / 64);
    k_gemm_wmma<<<grid, 128, 0, stream>>>(aggbf, hnbf, wbf1, wbf2, bias, hcur, N, Cin, Cout);
  };

  // ---- DOWN0: 3 -> 64 (scalar GEMM, Cin too small for WMMA) ----
  {
    const float* const* p = &P[0];
    run_pre(x, 3, nullptr, 0, p);
    run_scatter(hn, 3);
    k_gemm_scalar<<<gblk((size_t)N * 64), ELT_BLOCK, 0, stream>>>(agg, hn, rdeg, p[6], p[7], p[8],
                                                                  hcur, N, 3, 64, 1);
    hipMemcpyAsync(skip0, hcur, (size_t)N * 64 * 4, hipMemcpyDeviceToDevice, stream);
  }
  // ---- DOWN1: 64 -> 128 ----
  {
    const float* const* p = &P[9];
    run_pre(hcur, 64, nullptr, 0, p);
    run_scatter(hn, 64);
    run_gemm(p[6], p[8], p[7], 64, 128, true);
    hipMemcpyAsync(skip1, hcur, (size_t)N * 128 * 4, hipMemcpyDeviceToDevice, stream);
  }
  // ---- DOWN2: 128 -> 256 ----
  {
    const float* const* p = &P[18];
    run_pre(hcur, 128, nullptr, 0, p);
    run_scatter(hn, 128);
    run_gemm(p[6], p[8], p[7], 128, 256, true);
  }
  // ---- UP0: 256 -> 128 (add aggregation) ----
  {
    const float* const* p = &P[27];
    run_pre(hcur, 256, nullptr, 0, p);
    run_scatter(hn, 256);
    run_gemm(p[6], p[8], p[7], 256, 128, false);
  }
  // ---- UP1: concat(128,128)=256 -> 64 ----
  {
    const float* const* p = &P[36];
    run_pre(hcur, 128, skip1, 128, p);
    run_scatter(hn, 256);
    run_gemm(p[6], p[8], p[7], 256, 64, false);
  }
  // ---- UP2: concat(64,64)=128 -> 64 ----
  {
    const float* const* p = &P[45];
    run_pre(hcur, 64, skip0, 64, p);
    run_scatter(hn, 128);
    run_gemm(p[6], p[8], p[7], 128, 64, false);
  }
  // ---- LAST: GraphConv 64 -> 3, no LN/te, no activation ----
  {
    const float* const* p = &P[54];
    run_scatter(hcur, 64);
    k_gemm_scalar<<<gblk((size_t)N * 3), ELT_BLOCK, 0, stream>>>(agg, hcur, nullptr, p[0], p[1], p[2],
                                                                 (float*)d_out, N, 64, 3, 0);
  }
}
